// MultiHeadAttention_43585328120478
// MI455X (gfx1250) — matrix operand order repaired and checked by reading, not compiled, not run on hardware
//
#include <hip/hip_runtime.h>
#include <hip/hip_bf16.h>
#include <math.h>

// Problem constants
#define B_  16
#define C_  256
#define S_  4096          // H*W
#define NH_ 4
#define DH_ 64
#define TS_ 64            // seq positions per workgroup

typedef _Float16 v16h __attribute__((ext_vector_type(16)));
typedef _Float16 v8h  __attribute__((ext_vector_type(8)));
typedef float    v8f  __attribute__((ext_vector_type(8)));
typedef float    v4f  __attribute__((ext_vector_type(4)));

// LDS strides (elements). XSTR: f16, multiple of 8 so rows are 16B-aligned for b128.
#define XSTR 264          // 256 + 8   (Xs tile / attn-out tile, f16)
#define YSTR 772          // 768 + 4   (qkv tile, f16)
#define OSTR 68           // 64 + 4    (transposed out tile, f32; 68*4 % 16 == 0)
#define LDS_BYTES (64*XSTR*2 + 64*YSTR*2)   // 33792 + 98816 = 132608

// ---------------------------------------------------------------------------
// Weight pre-pack: fp32 -> f16 in WMMA B-fragment order.
// Fragment (nt, ks): 32 lanes x 16 f16 contiguous (32B per lane).
// B order used here (contiguous k): lanes 0-15 -> n=lane, K = ks*32 + 0..15 sequential;
//                           lanes 16-31 -> n=lane-16, K = ks*32 + 16..31.
// ---------------------------------------------------------------------------
__global__ void pack_weights_kernel(const float* __restrict__ wq,
                                    const float* __restrict__ wkv,
                                    const float* __restrict__ wo,
                                    _Float16* __restrict__ wqkv_p,
                                    _Float16* __restrict__ wo_p) {
  int idx = blockIdx.x * 256 + threadIdx.x;
  const int QKV_TOT = 48 * 8 * 32 * 16;   // 48 n-tiles x 8 k-steps x 512 f16
  if (idx < QKV_TOT) {
    int j    = idx & 15;
    int lane = (idx >> 4) & 31;
    int ks   = (idx >> 9) & 7;
    int nt   = idx >> 12;
    int n = nt * 16 + (lane & 15);
    int k = ks * 32 + ((lane >> 4) << 4) + j;
    float v = (n < C_) ? wq[k * C_ + n] : wkv[k * (2 * C_) + (n - C_)];
    wqkv_p[idx] = (_Float16)v;
  } else {
    int i2   = idx - QKV_TOT;             // 16 n-tiles x 8 k-steps x 512 f16
    int j    = i2 & 15;
    int lane = (i2 >> 4) & 31;
    int ks   = (i2 >> 9) & 7;
    int nt   = i2 >> 12;
    int n = nt * 16 + (lane & 15);
    int k = ks * 32 + ((lane >> 4) << 4) + j;
    wo_p[i2] = (_Float16)wo[k * C_ + n];
  }
}

// ---------------------------------------------------------------------------
// One 16-wide column block over K=256: acc[4] = A(64x256, LDS f16) x B(256x16).
// A order used here (contiguous k, as packed B): lanes 0-15: K in kb..kb+15, kb=ks*32;
//                            lanes 16-31: same but kb=ks*32+16.
// ---------------------------------------------------------------------------
static __device__ __forceinline__ void wmma_block(
    const _Float16* As,                 // LDS tile [64][XSTR]
    const _Float16* __restrict__ Bp,    // packed weights for this n-tile
    int lane, v8f acc[4]) {
  const v8f zero = {0.f, 0.f, 0.f, 0.f, 0.f, 0.f, 0.f, 0.f};
  acc[0] = zero; acc[1] = zero; acc[2] = zero; acc[3] = zero;
  const int lmod = lane & 15;
  const int koff = ((lane >> 4) & 1) * 16;
#pragma unroll
  for (int ks = 0; ks < 8; ++ks) {
    v16h bfrag = *(const v16h*)(Bp + (size_t)(ks * 32 + lane) * 16);
    int kb = ks * 32 + koff;
#pragma unroll
    for (int mt = 0; mt < 4; ++mt) {
      const _Float16* ap = As + (mt * 16 + lmod) * XSTR + kb;
      v8h lo = *(const v8h*)ap;          // ds_load_b128: K kb..kb+7
      v8h hi = *(const v8h*)(ap + 8);    // ds_load_b128: K kb+8..kb+15
      v16h afrag;
#pragma unroll
      for (int jj = 0; jj < 8; ++jj) { afrag[jj] = lo[jj]; afrag[8 + jj] = hi[jj]; }
      acc[mt] = __builtin_amdgcn_wmma_f32_16x16x32_f16(
          false, afrag, false, bfrag, (short)0, acc[mt], false, false);
    }
  }
}

// ---------------------------------------------------------------------------
// Fully fused: x-tile -> qkv gemm -> per-position head attention -> out gemm
// -> transposed coalesced store. 1024 blocks x 256 threads (8 waves).
// ---------------------------------------------------------------------------
__global__ __launch_bounds__(256) void mha_fused_kernel(
    const float* __restrict__ x,
    const float* __restrict__ bq,
    const float* __restrict__ bkv,
    const float* __restrict__ bo,
    const _Float16* __restrict__ wqkv_p,
    const _Float16* __restrict__ wo_p,
    float* __restrict__ out) {
  extern __shared__ char smem[];
  _Float16* Xs = (_Float16*)smem;                   // [64][XSTR]; reused as attn-out
  _Float16* Yq = (_Float16*)(smem + 64 * XSTR * 2); // [64][YSTR] qkv (f16)
  float*    oT = (float*)(smem + 64 * XSTR * 2);    // aliases Yq: [256][OSTR]

  const int t    = threadIdx.x;
  const int lane = t & 31;
  const int w    = t >> 5;
  const int b    = blockIdx.x >> 6;
  const int s0   = (blockIdx.x & 63) * TS_;

  // Phase 1: stage x tile -> LDS f16. x is [b][c][s] with s contiguous, so each
  // thread streams a float4 along s (global_load_b128) and transpose-stores
  // four f16 values into LDS rows sl4..sl4+3 at column c.
  const float* xb = x + (size_t)b * C_ * S_ + s0;
#pragma unroll 1
  for (int it = 0; it < 16; ++it) {
    int e   = it * 256 + t;      // 0..4095
    int sl4 = (e & 15) * 4;      // 0,4,..,60
    int c   = e >> 4;            // 0..255
    v4f v = *(const v4f*)(xb + (size_t)c * S_ + sl4);
    Xs[(sl4 + 0) * XSTR + c] = (_Float16)v.x;
    Xs[(sl4 + 1) * XSTR + c] = (_Float16)v.y;
    Xs[(sl4 + 2) * XSTR + c] = (_Float16)v.z;
    Xs[(sl4 + 3) * XSTR + c] = (_Float16)v.w;
  }
  __syncthreads();

  // Phase 2: qkv = Xs(64x256) @ Wqkv(256x768) + bias -> Yq (f16)
  {
    const int lmod  = lane & 15;
    const int mhalf = ((lane >> 4) & 1) * 8;   // C-layout: lanes>=16 hold M=r+8
    for (int i = 0; i < 6; ++i) {
      int nt = w * 6 + i;
      v8f acc[4];
      wmma_block(Xs, wqkv_p + (size_t)nt * 8 * 32 * 16, lane, acc);
      int n = nt * 16 + lmod;
      float bias = (n < C_) ? bq[n] : bkv[n - C_];
#pragma unroll
      for (int mt = 0; mt < 4; ++mt)
#pragma unroll
        for (int r = 0; r < 8; ++r)
          Yq[(mt * 16 + mhalf + r) * YSTR + n] = (_Float16)(acc[mt][r] + bias);
    }
  }
  __syncthreads();

  // Phase 3: per-position attention over heads (4x4 softmax, DH=64)
  {
    const int s  = t >> 2;   // 0..63
    const int nh = t & 3;    // head
    const _Float16* row = Yq + s * YSTR;
    const _Float16* qp  = row + nh * DH_;
    float sc[NH_];
#pragma unroll
    for (int m = 0; m < NH_; ++m) {
      const _Float16* kp = row + C_ + m * DH_;
      float dot = 0.f;
#pragma unroll 8
      for (int d = 0; d < DH_; ++d) dot += (float)qp[d] * (float)kp[d];
      sc[m] = dot * 0.125f;  // 1/sqrt(64)
    }
    float mx = fmaxf(fmaxf(sc[0], sc[1]), fmaxf(sc[2], sc[3]));
    float e0 = __expf(sc[0] - mx), e1 = __expf(sc[1] - mx);
    float e2 = __expf(sc[2] - mx), e3 = __expf(sc[3] - mx);
    float inv = 1.f / (e0 + e1 + e2 + e3);
    float a0 = e0 * inv, a1 = e1 * inv, a2 = e2 * inv, a3 = e3 * inv;
    const _Float16* vp = row + 2 * C_;
    _Float16* y2 = Xs + s * XSTR + nh * DH_;   // reuse Xs region as GEMM2 A-tile
#pragma unroll 8
    for (int d = 0; d < DH_; ++d) {
      float o = a0 * (float)vp[d] + a1 * (float)vp[DH_ + d]
              + a2 * (float)vp[2 * DH_ + d] + a3 * (float)vp[3 * DH_ + d];
      y2[d] = (_Float16)o;
    }
  }
  __syncthreads();

  // Phase 4: out = attn(64x256) @ Wo(256x256) + bo -> oT transposed in LDS
  {
    const int lmod  = lane & 15;
    const int mhalf = ((lane >> 4) & 1) * 8;
    for (int i = 0; i < 2; ++i) {
      int nt = w * 2 + i;
      v8f acc[4];
      wmma_block(Xs, wo_p + (size_t)nt * 8 * 32 * 16, lane, acc);
      int n = nt * 16 + lmod;
      float bias = bo[n];
#pragma unroll
      for (int mt = 0; mt < 4; ++mt)
#pragma unroll
        for (int r = 0; r < 8; ++r)
          oT[n * OSTR + (mt * 16 + mhalf + r)] = acc[mt][r] + bias;
    }
  }
  __syncthreads();

  // Phase 5: coalesced store out[b][c][s0..s0+63] (16 lanes -> 256B contiguous)
  float* ob = out + (size_t)b * C_ * S_ + s0;
#pragma unroll 1
  for (int cb = 0; cb < 16; ++cb) {
    int c   = cb * 16 + (t >> 4);
    int sl4 = (t & 15) * 4;
    v4f val = *(const v4f*)(oT + c * OSTR + sl4);
    *(v4f*)(ob + (size_t)c * S_ + sl4) = val;
  }
}

extern "C" void kernel_launch(void* const* d_in, const int* in_sizes, int n_in,
                              void* d_out, int out_size, void* d_ws, size_t ws_size,
                              hipStream_t stream) {
  (void)in_sizes; (void)n_in; (void)out_size; (void)ws_size;
  const float* x   = (const float*)d_in[0];
  const float* wq  = (const float*)d_in[1];
  const float* bq  = (const float*)d_in[2];
  const float* wkv = (const float*)d_in[3];
  const float* bkv = (const float*)d_in[4];
  const float* wo  = (const float*)d_in[5];
  const float* bo  = (const float*)d_in[6];
  float* out = (float*)d_out;

  // Workspace: packed f16 weights (384 KiB + 128 KiB)
  _Float16* wqkv_p = (_Float16*)d_ws;
  _Float16* wo_p   = (_Float16*)((char*)d_ws + (size_t)48 * 8 * 32 * 16 * 2);

  pack_weights_kernel<<<1024, 256, 0, stream>>>(wq, wkv, wo, wqkv_p, wo_p);

  (void)hipFuncSetAttribute((const void*)mha_fused_kernel,
                            hipFuncAttributeMaxDynamicSharedMemorySize,
                            (int)LDS_BYTES);
  // 16 batches x 64 seq-tiles = 1024 workgroups, 256 threads (8 wave32)
  mha_fused_kernel<<<1024, 256, LDS_BYTES, stream>>>(x, bq, bkv, bo,
                                                     wqkv_p, wo_p, out);
}
